// CoAttenBlock_16234976379261
// MI455X (gfx1250) — matrix operand order repaired and checked by reading, not compiled, not run on hardware
//
#include <hip/hip_runtime.h>

#define DEVFN __device__ __forceinline__

typedef _Float16 f16;
typedef _Float16 __attribute__((ext_vector_type(16))) v16h;
typedef _Float16 __attribute__((ext_vector_type(8)))  v8h;
typedef float    __attribute__((ext_vector_type(8)))  v8f;

static constexpr int B_  = 8;
static constexpr int C_  = 64;
static constexpr int C2_ = 128;
static constexpr int HW_ = 2304;                    // 48*48
static constexpr long long HWHW_ = (long long)HW_ * HW_;
static constexpr int NCH_ = 18;                     // column-reduce chunks (2304/128)

DEVFN v16h cat8(v8h lo, v8h hi) {
    return __builtin_shufflevector(lo, hi, 0,1,2,3,4,5,6,7,8,9,10,11,12,13,14,15);
}

// A-matrix fragment load (16x32 f16): lanes 0-15 hold K {0..15},
// lanes 16-31 hold K {16..31} of their M row (contiguous, matches B fills).
DEVFN v16h load_a16(const f16* rowbase_k0, int lane) {
    const int k0 = (lane >> 4) * 16;
    v8h lo = *(const v8h*)(rowbase_k0 + k0);
    v8h hi = *(const v8h*)(rowbase_k0 + k0 + 8);
    return cat8(lo, hi);
}

DEVFN v8f wmma_f16(v16h a, v16h b, v8f c) {
    return __builtin_amdgcn_wmma_f32_16x16x32_f16(false, a, false, b, (short)0, c, false, false);
}

// ---------------------------------------------------------------------------
// Kernel 1: input 1x1 convs. Produces fp32 xL/xR plus f16 copies in [c][n]
// and transposed [n][c] layouts (for WMMA A/B fragment contiguity).
// grid: (B*C*HW/256, 2 sides), block 256. Each block = one (b,c), 256 n's
// => weight loads are wave-uniform (scalar cache path).
// ---------------------------------------------------------------------------
__global__ void k_conv_in(const float* __restrict__ xlh, const float* __restrict__ xll,
                          const float* __restrict__ xrh, const float* __restrict__ xrl,
                          const float* __restrict__ wL,  const float* __restrict__ bL,
                          const float* __restrict__ wR,  const float* __restrict__ bR,
                          float* __restrict__ xLf, float* __restrict__ xRf,
                          f16* __restrict__ xLh,  f16* __restrict__ xRh,
                          f16* __restrict__ xLhT, f16* __restrict__ xRhT)
{
    const int side = blockIdx.y;
    const float* xa   = side ? xrh : xlh;
    const float* xb   = side ? xrl : xll;
    const float* w    = side ? wR  : wL;
    const float* bias = side ? bR  : bL;
    float* xf  = side ? xRf  : xLf;
    f16*   xh  = side ? xRh  : xLh;
    f16*   xhT = side ? xRhT : xLhT;

    const int idx = blockIdx.x * blockDim.x + threadIdx.x;    // b*C*HW, n fastest
    const int n = idx % HW_;
    const int c = (idx / HW_) % C_;
    const int b = idx / (HW_ * C_);

    const float* wrow = w + c * C2_;
    const float* pa = xa + (long long)b * C_ * HW_ + n;
    const float* pb = xb + (long long)b * C_ * HW_ + n;
    float acc = bias[c];
#pragma unroll 8
    for (int j = 0; j < C_; ++j) {
        acc += wrow[j]      * pa[(long long)j * HW_];
        acc += wrow[C_ + j] * pb[(long long)j * HW_];
    }
    const long long o = ((long long)b * C_ + c) * HW_ + n;
    xf[o] = acc;
    const f16 h = (f16)acc;
    xh[o] = h;
    xhT[((long long)b * HW_ + n) * C_ + c] = h;
}

// ---------------------------------------------------------------------------
// Kernel 2: aff[n,m] = sum_c xL[c,n]*xR[c,m].  WMMA f16 GEMM, K=64.
// Each wave computes a 32x32 block (2x2 fragments). grid (36,36,NB), block 128.
// ---------------------------------------------------------------------------
__global__ void k_aff(const f16* __restrict__ xLhT, const f16* __restrict__ xRhT,
                      float* __restrict__ aff, int bBase)
{
    const int b = bBase + blockIdx.z;
    float* affb = aff + (long long)blockIdx.z * HWHW_;
    const int lane = threadIdx.x & 31;
    const int w    = threadIdx.x >> 5;                 // 0..3
    const int nb0  = blockIdx.x * 64 + (w >> 1) * 32;
    const int mb0  = blockIdx.y * 64 + (w & 1) * 32;
    const f16* Ab = xLhT + (long long)b * HW_ * C_;    // [n][c]
    const f16* Bb = xRhT + (long long)b * HW_ * C_;    // [m][c]

    v8f acc[2][2] = {};
#pragma unroll
    for (int kc = 0; kc < C_; kc += 32) {
        v16h a[2], bm[2];
#pragma unroll
        for (int t = 0; t < 2; ++t) {
            const int row = nb0 + t * 16 + (lane & 15);
            a[t] = load_a16(Ab + (long long)row * C_ + kc, lane);
            const int col = mb0 + t * 16 + (lane & 15);
            bm[t] = *(const v16h*)(Bb + (long long)col * C_ + kc + (lane >> 4) * 16);
        }
#pragma unroll
        for (int i = 0; i < 2; ++i)
#pragma unroll
            for (int j = 0; j < 2; ++j)
                acc[i][j] = wmma_f16(a[i], bm[j], acc[i][j]);
    }
#pragma unroll
    for (int i = 0; i < 2; ++i)
#pragma unroll
        for (int j = 0; j < 2; ++j) {
            const int row0 = nb0 + i * 16 + (lane >> 4) * 8;
            const int col  = mb0 + j * 16 + (lane & 15);
#pragma unroll
            for (int v = 0; v < 8; ++v)
                affb[(long long)(row0 + v) * HW_ + col] = acc[i][j][v];
        }
}

// ---------------------------------------------------------------------------
// Kernel 3a: partial column (axis=1) softmax stats over 128-row chunks.
// grid (18, 9, NB), block 256. Online max/sum; coalesced reads, L2-resident.
// ---------------------------------------------------------------------------
__global__ void k_colpart(const float* __restrict__ aff,
                          float* __restrict__ pmax, float* __restrict__ psum, int bBase)
{
    const float* affb = aff + (long long)blockIdx.z * HWHW_;
    const int m  = blockIdx.y * blockDim.x + threadIdx.x;
    const int ch = blockIdx.x;
    const int n0 = ch * 128;
    float mx = -3.0e38f, s = 0.f;
    for (int i = 0; i < 128; ++i) {
        const float v = affb[(long long)(n0 + i) * HW_ + m];
        const float nm = fmaxf(mx, v);
        s = s * __expf(mx - nm) + __expf(v - nm);
        mx = nm;
    }
    const int b = bBase + blockIdx.z;
    const long long o = ((long long)b * NCH_ + ch) * HW_ + m;
    pmax[o] = mx;  psum[o] = s;
}

// Kernel 3b: combine partials -> colmax, 1/colsum. grid (9,1,NB), block 256.
__global__ void k_colfin(const float* __restrict__ pmax, const float* __restrict__ psum,
                         float* __restrict__ colmax, float* __restrict__ colsuminv, int bBase)
{
    const int m = blockIdx.x * blockDim.x + threadIdx.x;
    const int b = bBase + blockIdx.z;
    const float* pm = pmax + (long long)b * NCH_ * HW_ + m;
    const float* ps = psum + (long long)b * NCH_ * HW_ + m;
    float mx = -3.0e38f;
    for (int ch = 0; ch < NCH_; ++ch) mx = fmaxf(mx, pm[(long long)ch * HW_]);
    float s = 0.f;
    for (int ch = 0; ch < NCH_; ++ch)
        s += ps[(long long)ch * HW_] * __expf(pm[(long long)ch * HW_] - mx);
    colmax[(long long)b * HW_ + m]    = mx;
    colsuminv[(long long)b * HW_ + m] = 1.0f / s;
}

// Kernel 3c: row (axis=2) softmax stats, one wave per row + shfl butterfly.
// grid (288,1,NB), block 256 (8 waves).
__global__ void k_row(const float* __restrict__ aff,
                      float* __restrict__ rowmax, float* __restrict__ rowsuminv, int bBase)
{
    const float* affb = aff + (long long)blockIdx.z * HWHW_;
    const int lane = threadIdx.x & 31;
    const int n = blockIdx.x * 8 + (threadIdx.x >> 5);
    const float* row = affb + (long long)n * HW_;
    float mx = -3.0e38f, s = 0.f;
    for (int m = lane; m < HW_; m += 32) {
        const float v = row[m];
        const float nm = fmaxf(mx, v);
        s = s * __expf(mx - nm) + __expf(v - nm);
        mx = nm;
    }
#pragma unroll
    for (int off = 16; off > 0; off >>= 1) {
        const float omx = __shfl_xor(mx, off, 32);
        const float os  = __shfl_xor(s,  off, 32);
        const float nm = fmaxf(mx, omx);
        s = s * __expf(mx - nm) + os * __expf(omx - nm);
        mx = nm;
    }
    if (lane == 0) {
        const int b = bBase + blockIdx.z;
        rowmax[(long long)b * HW_ + n]    = mx;
        rowsuminv[(long long)b * HW_ + n] = 1.0f / s;
    }
}

// Zero b1/b2 accumulators (k_attn uses fp atomics). grid (576, NB), block 256.
__global__ void k_zero(float* __restrict__ b1, float* __restrict__ b2, int bBase)
{
    const int b = bBase + blockIdx.y;
    const long long o = (long long)b * C_ * HW_ + blockIdx.x * blockDim.x + threadIdx.x;
    b1[o] = 0.f;  b2[o] = 0.f;
}

// ---------------------------------------------------------------------------
// Kernel 4: fused double-attention aggregation.
//   b1[c,m] = (sum_n xL[c,n] * exp(aff[n,m]-colmax[m])) / colsum[m]
//   b2[c,m] =  sum_n xR[c,n] * exp(aff[n,m]-rowmax[n]) / rowsum[n]
// One wave: 16-wide m-tile, all 64 c, both branches (8 accumulators),
// split-K=4 over n with hardware global_atomic_add_f32 reduction.
// grid (144,1,NB), block 128 (4 waves => 576 waves/batch).
// ---------------------------------------------------------------------------
__global__ void __launch_bounds__(128) k_attn(
    const float* __restrict__ aff,
    const f16* __restrict__ xLh, const f16* __restrict__ xRh,
    const float* __restrict__ colmax, const float* __restrict__ colsuminv,
    const float* __restrict__ rowmax, const float* __restrict__ rowsuminv,
    float* __restrict__ b1, float* __restrict__ b2, int bBase)
{
    const int b = bBase + blockIdx.z;
    const float* affb = aff + (long long)blockIdx.z * HWHW_;
    const int lane = threadIdx.x & 31;
    const int gw = blockIdx.x * 4 + (threadIdx.x >> 5);   // 0..575
    const int mb = (gw % 144) * 16;
    const int ks = gw / 144;                              // split-K id 0..3
    const int m  = mb + (lane & 15);
    const int kh = lane >> 4;

    const float cm  = colmax[(long long)b * HW_ + m];
    const float csi = colsuminv[(long long)b * HW_ + m];
    const float* rm  = rowmax    + (long long)b * HW_;
    const float* rsi = rowsuminv + (long long)b * HW_;
    const f16* AL = xLh + (long long)b * C_ * HW_;        // [c][n]
    const f16* AR = xRh + (long long)b * C_ * HW_;

    v8f acc1[4] = {}, acc2[4] = {};

    const int n0 = ks * (HW_ / 4);
    for (int nc = n0; nc < n0 + HW_ / 4; nc += 32) {
        // Build both probability B-fragments from one aff read.
        const int nlane = nc + kh * 16;
        v16h e1v, e2v;
#pragma unroll
        for (int kk = 0; kk < 16; ++kk) {
            const int n = nlane + kk;
            const float v = affb[(long long)n * HW_ + m];
            e1v[kk] = (f16)__expf(v - cm);
            e2v[kk] = (f16)(__expf(v - rm[n]) * rsi[n]);
        }
#pragma unroll
        for (int ct = 0; ct < 4; ++ct) {
            const int c = ct * 16 + (lane & 15);
            v16h aL = load_a16(AL + (long long)c * HW_ + nc, lane);
            acc1[ct] = wmma_f16(aL, e1v, acc1[ct]);
            v16h aR = load_a16(AR + (long long)c * HW_ + nc, lane);
            acc2[ct] = wmma_f16(aR, e2v, acc2[ct]);
        }
    }
#pragma unroll
    for (int ct = 0; ct < 4; ++ct) {
#pragma unroll
        for (int v = 0; v < 8; ++v) {
            const int c = ct * 16 + v + kh * 8;
            const long long o = ((long long)b * C_ + c) * HW_ + m;
            unsafeAtomicAdd(&b1[o], acc1[ct][v] * csi);
            unsafeAtomicAdd(&b2[o], acc2[ct][v]);
        }
    }
}

// Kernel 5: sigmoid gates (K=64 dot per pixel). grid (9,1,NB), block 256.
__global__ void k_gate(const float* __restrict__ b1, const float* __restrict__ b2,
                       const float* __restrict__ gLw, const float* __restrict__ gLb,
                       const float* __restrict__ gRw, const float* __restrict__ gRb,
                       float* __restrict__ g1, float* __restrict__ g2, int bBase)
{
    const int m = blockIdx.x * blockDim.x + threadIdx.x;
    const int b = bBase + blockIdx.z;
    float a1 = gLb[0], a2 = gRb[0];
    const float* p1 = b1 + (long long)b * C_ * HW_ + m;
    const float* p2 = b2 + (long long)b * C_ * HW_ + m;
#pragma unroll 8
    for (int c = 0; c < C_; ++c) {
        a1 += gLw[c] * p1[(long long)c * HW_];
        a2 += gRw[c] * p2[(long long)c * HW_];
    }
    g1[(long long)b * HW_ + m] = 1.f / (1.f + __expf(-a1));
    g2[(long long)b * HW_ + m] = 1.f / (1.f + __expf(-a2));
}

// Kernel 6: output convs out = W[:, :64]@x + W[:, 64:]@(b*g) + bias.
// grid (9, 128, NB): y = side*64 + c_out (weights wave-uniform). block 256.
__global__ void k_outconv(const float* __restrict__ xLf, const float* __restrict__ xRf,
                          const float* __restrict__ b1,  const float* __restrict__ b2,
                          const float* __restrict__ g1,  const float* __restrict__ g2,
                          const float* __restrict__ wLo, const float* __restrict__ bLo,
                          const float* __restrict__ wRo, const float* __restrict__ bRo,
                          float* __restrict__ out, int bBase)
{
    const int side = blockIdx.y >> 6;
    const int c    = blockIdx.y & 63;
    const int m = blockIdx.x * blockDim.x + threadIdx.x;
    const int b = bBase + blockIdx.z;
    const float* x  = side ? xRf : xLf;
    const float* bb = side ? b2  : b1;
    const float* g  = side ? g2  : g1;
    const float* w  = side ? wRo : wLo;
    const float* bi = side ? bRo : bLo;
    const float* wrow = w + c * C2_;
    const float gv = g[(long long)b * HW_ + m];
    const float* px = x  + (long long)b * C_ * HW_ + m;
    const float* pb = bb + (long long)b * C_ * HW_ + m;
    float acc = bi[c];
#pragma unroll 8
    for (int j = 0; j < C_; ++j) {
        acc += wrow[j]      * px[(long long)j * HW_];
        acc += wrow[C_ + j] * (pb[(long long)j * HW_] * gv);
    }
    out[(long long)side * B_ * C_ * HW_ + ((long long)b * C_ + c) * HW_ + m] = acc;
}

// ---------------------------------------------------------------------------
extern "C" void kernel_launch(void* const* d_in, const int* in_sizes, int n_in,
                              void* d_out, int out_size, void* d_ws, size_t ws_size,
                              hipStream_t stream)
{
    const float* xlh = (const float*)d_in[0];
    const float* xll = (const float*)d_in[1];
    const float* xrh = (const float*)d_in[2];
    const float* xrl = (const float*)d_in[3];
    const float* wL  = (const float*)d_in[4];
    const float* bL  = (const float*)d_in[5];
    const float* wR  = (const float*)d_in[6];
    const float* bR  = (const float*)d_in[7];
    const float* gLw = (const float*)d_in[8];
    const float* gLb = (const float*)d_in[9];
    const float* gRw = (const float*)d_in[10];
    const float* gRb = (const float*)d_in[11];
    const float* wLo = (const float*)d_in[12];
    const float* bLo = (const float*)d_in[13];
    const float* wRo = (const float*)d_in[14];
    const float* bRo = (const float*)d_in[15];
    float* out = (float*)d_out;

    char* ws = (char*)d_ws;
    auto alloc = [&](size_t bytes) -> char* {
        char* p = ws;  ws += (bytes + 255) & ~size_t(255);  return p;
    };
    const size_t fBCHW = (size_t)B_ * C_ * HW_;
    float* xLf  = (float*)alloc(fBCHW * 4);
    float* xRf  = (float*)alloc(fBCHW * 4);
    f16*   xLh  = (f16*)  alloc(fBCHW * 2);
    f16*   xRh  = (f16*)  alloc(fBCHW * 2);
    f16*   xLhT = (f16*)  alloc(fBCHW * 2);
    f16*   xRhT = (f16*)  alloc(fBCHW * 2);
    float* b1   = (float*)alloc(fBCHW * 4);
    float* b2   = (float*)alloc(fBCHW * 4);
    float* g1   = (float*)alloc((size_t)B_ * HW_ * 4);
    float* g2   = (float*)alloc((size_t)B_ * HW_ * 4);
    float* colmax    = (float*)alloc((size_t)B_ * HW_ * 4);
    float* colsuminv = (float*)alloc((size_t)B_ * HW_ * 4);
    float* rowmax    = (float*)alloc((size_t)B_ * HW_ * 4);
    float* rowsuminv = (float*)alloc((size_t)B_ * HW_ * 4);
    float* pmax = (float*)alloc((size_t)B_ * NCH_ * HW_ * 4);
    float* psum = (float*)alloc((size_t)B_ * NCH_ * HW_ * 4);
    const size_t fixed = (size_t)(ws - (char*)d_ws);
    // aff: all 8 batches (170 MB) if ws allows -> fully parallel;
    // otherwise one 21 MB batch buffer, serialized chain per batch.
    const size_t affBatchBytes = (size_t)HWHW_ * 4;
    const int NB = (ws_size >= fixed + (size_t)B_ * affBatchBytes) ? B_ : 1;
    float* aff = (float*)ws;

    k_conv_in<<<dim3(fBCHW / 256, 2), 256, 0, stream>>>(
        xlh, xll, xrh, xrl, wL, bL, wR, bR, xLf, xRf, xLh, xRh, xLhT, xRhT);

    for (int b0 = 0; b0 < B_; b0 += NB) {
        k_aff    <<<dim3(36, 36, NB), 128, 0, stream>>>(xLhT, xRhT, aff, b0);
        k_colpart<<<dim3(NCH_, 9, NB), 256, 0, stream>>>(aff, pmax, psum, b0);
        k_colfin <<<dim3(9, 1, NB), 256, 0, stream>>>(pmax, psum, colmax, colsuminv, b0);
        k_row    <<<dim3(288, 1, NB), 256, 0, stream>>>(aff, rowmax, rowsuminv, b0);
        k_zero   <<<dim3(576, NB, 1), 256, 0, stream>>>(b1, b2, b0);
        k_attn   <<<dim3(144, 1, NB), 128, 0, stream>>>(
            aff, xLh, xRh, colmax, colsuminv, rowmax, rowsuminv, b1, b2, b0);
        k_gate   <<<dim3(9, 1, NB), 256, 0, stream>>>(b1, b2, gLw, gLb, gRw, gRb, g1, g2, b0);
        k_outconv<<<dim3(9, 128, NB), 256, 0, stream>>>(
            xLf, xRf, b1, b2, g1, g2, wLo, bLo, wRo, bRo, out, b0);
    }
    (void)in_sizes; (void)n_in; (void)out_size;
}
